// MaGNet_69501160784161
// MI455X (gfx1250) — matrix operand order repaired and checked by reading, not compiled, not run on hardware
//
#include <hip/hip_runtime.h>
#include <hip/hip_bf16.h>

// ---------------- model constants ----------------
constexpr int Bc = 4, Nc = 207, Tc = 96, Fc = 8, Dc = 32;
constexpr int Ec = 128, Sc = 16;           // mamba expand dim, d_state
constexpr int BNc = Bc * Nc;               // 828 sequences
constexpr int Rr  = BNc * Tc;              // 79488 token rows
constexpr int NTc = Nc * Tc;               // 19872 rows per batch
constexpr int TD  = Tc * Dc;               // 3072

// output layout (flat float32, reference tuple order)
constexpr long OZ     = 0;
constexpr long OATT   = OZ + (long)Bc * Nc * 2;                  // 1656
constexpr long OTOPK  = OATT + (long)BNc * Tc * Tc;              // +7630848
constexpr long OHLOC  = OTOPK + (long)Rr * 8;                    // +635904
constexpr long OHG    = OHLOC + (long)Rr * 16;                   // +1271808
constexpr long OWEDGE = OHG + (long)Rr * 16;                     // +1271808

// ---------------- WMMA types ----------------
typedef __attribute__((ext_vector_type(16))) _Float16 v16h;
typedef __attribute__((ext_vector_type(8)))  float    v8f;

// =====================================================================
// Generic GEMM: C[M,N] = epilogue(A[M,K] @ B[K,N])
// Requirements (all call sites satisfy them): K % 32 == 0, N % 16 == 0,
// lda % 4 == 0, A 16-byte aligned. M may be ragged (clamp + store guard).
// One wave32 computes a 64x16 tile (4 stacked 16x16 WMMAs sharing one B
// fragment). Inner loop is branch-free: EXEC stays all-ones for WMMA.
// epilogue: +bias[col]; act(0 none,1 relu,2 softplus); *rowScale[row*rss];
//           accum? C+=v : C=v
// =====================================================================
constexpr int MBLK = 4;

__global__ void wmma_gemm_kernel(const float* __restrict__ A, int lda,
                                 const float* __restrict__ B, int ldb,
                                 float* __restrict__ C, int ldc,
                                 int M, int N, int K,
                                 const float* __restrict__ bias, int act,
                                 const float* __restrict__ rowScale, int rss,
                                 int accum) {
  int wave = threadIdx.x >> 5;
  int lane = threadIdx.x & 31;
  int ntN  = N >> 4;
  int ntM  = (M + 15) >> 4;
  int ntMb = (ntM + MBLK - 1) / MBLK;
  int ntiles = ntMb * ntN;
  int tile = blockIdx.x * (blockDim.x >> 5) + wave;
  if (tile >= ntiles) return;                 // wave-uniform exit
  int tmb = tile / ntN, tn = tile % ntN;
  int l15 = lane & 15;
  int hi  = lane >> 4;                        // lane-half selector

  int colo = tn * 16 + l15;                   // true column (always < N)
  const float* Bcol = B + colo;

  // clamped A row offsets: out-of-range rows only feed discarded C rows
  long arow[MBLK];
#pragma unroll
  for (int f = 0; f < MBLK; ++f) {
    int r = tmb * (16 * MBLK) + f * 16 + l15;
    arow[f] = (long)(r < M ? r : (M - 1)) * lda;
  }

  v8f acc[MBLK];
#pragma unroll
  for (int f = 0; f < MBLK; ++f) acc[f] = v8f{};

  for (int k0 = 0; k0 < K; k0 += 32) {
    // ---- B fragment: 32x16 f16, lanes0-15 K=k0..k0+15, lanes16-31 +16 ----
    v16h b{};
    const float* bp = Bcol + (long)(k0 + (hi << 4)) * ldb;
#pragma unroll
    for (int i = 0; i < 16; ++i) b[i] = (_Float16)bp[(long)i * ldb];

    // ---- A fragments: per lane one contiguous 16-float run, b128 loads ----
#pragma unroll
    for (int f = 0; f < MBLK; ++f) {
      const float* ap = A + arow[f] + k0 + (hi << 4);
      float4 a0 = *(const float4*)(ap);
      float4 a1 = *(const float4*)(ap + 4);
      float4 a2 = *(const float4*)(ap + 8);
      float4 a3 = *(const float4*)(ap + 12);
      v16h a{};
      a[0]  = (_Float16)a0.x; a[1]  = (_Float16)a0.y;
      a[2]  = (_Float16)a0.z; a[3]  = (_Float16)a0.w;
      a[4]  = (_Float16)a1.x; a[5]  = (_Float16)a1.y;
      a[6]  = (_Float16)a1.z; a[7]  = (_Float16)a1.w;
      a[8]  = (_Float16)a2.x; a[9]  = (_Float16)a2.y;
      a[10] = (_Float16)a2.z; a[11] = (_Float16)a2.w;
      a[12] = (_Float16)a3.x; a[13] = (_Float16)a3.y;
      a[14] = (_Float16)a3.z; a[15] = (_Float16)a3.w;
      acc[f] = __builtin_amdgcn_wmma_f32_16x16x32_f16(false, a, false, b,
                                                      (short)0, acc[f],
                                                      false, false);
    }
  }

  float bv = bias ? bias[colo] : 0.f;
#pragma unroll
  for (int f = 0; f < MBLK; ++f) {
    int rbase = tmb * (16 * MBLK) + f * 16 + hi * 8;  // C/D row layout
#pragma unroll
    for (int r = 0; r < 8; ++r) {
      int rr = rbase + r;
      if (rr < M) {
        float v = acc[f][r] + bv;
        if (act == 1)      v = fmaxf(v, 0.f);
        else if (act == 2) v = (v > 20.f) ? v : log1pf(expf(v));
        if (rowScale) v *= rowScale[(long)rr * rss];
        long idx = (long)rr * ldc + colo;
        C[idx] = accum ? (C[idx] + v) : v;
      }
    }
  }
}

// ---------------- elementwise / small kernels ----------------
__global__ void zero_kernel(float* __restrict__ p, int total) {
  int i = blockIdx.x * blockDim.x + threadIdx.x;
  if (i < total) p[i] = 0.f;
}

__global__ void emb_kernel(const float* __restrict__ x, const float* __restrict__ W1,
                           const float* __restrict__ b1, const float* __restrict__ W2,
                           const float* __restrict__ b2, float* __restrict__ h, int rows) {
  int r = blockIdx.x * blockDim.x + threadIdx.x;
  if (r >= rows) return;
  float xin[Fc], hid[Fc];
#pragma unroll
  for (int i = 0; i < Fc; ++i) xin[i] = x[(long)r * Fc + i];
#pragma unroll
  for (int j = 0; j < Fc; ++j) {
    float s = b1[j];
#pragma unroll
    for (int i = 0; i < Fc; ++i) s += xin[i] * W1[i * Fc + j];
    hid[j] = fmaxf(s, 0.f);
  }
  for (int j = 0; j < Dc; ++j) {
    float s = b2[j];
#pragma unroll
    for (int i = 0; i < Fc; ++i) s += hid[i] * W2[i * Dc + j];
    h[(long)r * Dc + j] = fmaxf(s, 0.f);
  }
}

// residual + LayerNorm over D=32: one wave32 per row, shuffle reductions
__global__ void add_ln32_kernel(float* __restrict__ h, const float* __restrict__ resid,
                                const float* __restrict__ g, const float* __restrict__ bb,
                                int rows) {
  int wid = threadIdx.x >> 5, lane = threadIdx.x & 31;
  int row = blockIdx.x * (blockDim.x >> 5) + wid;
  if (row >= rows) return;
  float v = h[(long)row * 32 + lane];
  if (resid) v += resid[(long)row * 32 + lane];
  float s = v;
  for (int o = 16; o > 0; o >>= 1) s += __shfl_xor(s, o);
  float mean = s * (1.f / 32.f);
  float d = v - mean;
  float vs = d * d;
  for (int o = 16; o > 0; o >>= 1) vs += __shfl_xor(vs, o);
  float var = vs * (1.f / 32.f);
  h[(long)row * 32 + lane] = d * rsqrtf(var + 1e-5f) * g[lane] + bb[lane];
}

// mamba selective scan: one thread per (bn, e) channel
__global__ void mamba_scan_kernel(const float* __restrict__ xz, const float* __restrict__ dt,
                                  const float* __restrict__ Bm, const float* __restrict__ Cm,
                                  const float* __restrict__ Alog, const float* __restrict__ Dskip,
                                  float* __restrict__ y) {
  int idx = blockIdx.x * blockDim.x + threadIdx.x;
  if (idx >= BNc * Ec) return;
  int bn = idx / Ec, e = idx % Ec;
  float Aneg[Sc], st[Sc];
#pragma unroll
  for (int s = 0; s < Sc; ++s) { Aneg[s] = -expf(Alog[e * Sc + s]); st[s] = 0.f; }
  float dsk = Dskip[e];
  for (int t = 0; t < Tc; ++t) {
    long rt = (long)bn * Tc + t;
    float xm  = xz[rt * 256 + e];
    float zv  = xz[rt * 256 + Ec + e];
    float dtv = dt[rt * Ec + e];
    float dx  = dtv * xm;
    float accv = 0.f;
#pragma unroll
    for (int s = 0; s < Sc; ++s) {
      float dA = expf(dtv * Aneg[s]);
      st[s] = dA * st[s] + dx * Bm[rt * Sc + s];
      accv += st[s] * Cm[rt * Sc + s];
    }
    float yv = accv + xm * dsk;
    float sig = 1.f / (1.f + expf(-zv));
    y[rt * Ec + e] = yv * (zv * sig);
  }
}

// MoE softmax gate
__global__ void moe_gate_kernel(const float* __restrict__ h, const float* __restrict__ Wg,
                                float* __restrict__ gate, int rows) {
  int r = blockIdx.x * blockDim.x + threadIdx.x;
  if (r >= rows) return;
  float lg[4] = {0, 0, 0, 0};
  for (int d = 0; d < 32; ++d) {
    float hv = h[(long)r * 32 + d];
#pragma unroll
    for (int e = 0; e < 4; ++e) lg[e] += hv * Wg[d * 4 + e];
  }
  float m = fmaxf(fmaxf(lg[0], lg[1]), fmaxf(lg[2], lg[3]));
  float s = 0.f;
#pragma unroll
  for (int e = 0; e < 4; ++e) { lg[e] = expf(lg[e] - m); s += lg[e]; }
  float inv = 1.f / s;
#pragma unroll
  for (int e = 0; e < 4; ++e) gate[(long)r * 4 + e] = lg[e] * inv;
}

// GRU scan: one wave32 per sequence, Wh staged in LDS
__global__ void gru_scan_kernel(const float* __restrict__ gi, const float* __restrict__ Wh,
                                const float* __restrict__ bh, float* __restrict__ ys) {
  int bn = blockIdx.x;
  int lane = threadIdx.x;
  __shared__ float wls[32 * 96];
  __shared__ float hc[32];
  for (int i = lane; i < 32 * 96; i += 32) wls[i] = Wh[i];
  hc[lane] = 0.f;
  __syncthreads();
  for (int t = 0; t < Tc; ++t) {
    long rt = (long)bn * Tc + t;
    float ghr = bh[lane], ghz = bh[32 + lane], ghn = bh[64 + lane];
    for (int d = 0; d < 32; ++d) {
      float hv = hc[d];
      ghr += hv * wls[d * 96 + lane];
      ghz += hv * wls[d * 96 + 32 + lane];
      ghn += hv * wls[d * 96 + 64 + lane];
    }
    float ir = gi[rt * 96 + lane], iz = gi[rt * 96 + 32 + lane], inn = gi[rt * 96 + 64 + lane];
    float rr = 1.f / (1.f + expf(-(ir + ghr)));
    float zz = 1.f / (1.f + expf(-(iz + ghz)));
    float nn = tanhf(inn + rr * ghn);
    float nh = (1.f - zz) * nn + zz * hc[lane];
    __syncthreads();
    hc[lane] = nh;
    ys[rt * 32 + lane] = nh;
    __syncthreads();
  }
}

// flash-style attention, dh=8; block per (batch, head); K/V staged in LDS.
// optional amean: atomicAdd of prob/NH (buffer must be pre-zeroed).
__global__ void attn_kernel(const float* __restrict__ Q, const float* __restrict__ Kk,
                            const float* __restrict__ V, float* __restrict__ O,
                            int L, int NH, int causal, float* amean, float invnh) {
  const int DH = 8;
  int b = blockIdx.x / NH, h = blockIdx.x % NH;
  int Dm = NH * DH;
  __shared__ float kls[207 * 8];
  __shared__ float vls[207 * 8];
  for (int idx = threadIdx.x; idx < L * DH; idx += blockDim.x) {
    int j = idx / DH, d = idx % DH;
    kls[idx] = Kk[((long)b * L + j) * Dm + h * DH + d];
    vls[idx] = V[((long)b * L + j) * Dm + h * DH + d];
  }
  __syncthreads();
  const float rs = 0.35355339059327379f;  // 1/sqrt(8)
  for (int qi = threadIdx.x; qi < L; qi += blockDim.x) {
    float qr[8];
#pragma unroll
    for (int d = 0; d < 8; ++d) qr[d] = Q[((long)b * L + qi) * Dm + h * DH + d];
    float m = -3.0e38f, l = 0.f, acc[8] = {0, 0, 0, 0, 0, 0, 0, 0};
    int jend = causal ? (qi + 1) : L;
    for (int j = 0; j < jend; ++j) {
      float s = 0.f;
#pragma unroll
      for (int d = 0; d < 8; ++d) s += qr[d] * kls[j * 8 + d];
      s *= rs;
      float nm = fmaxf(m, s);
      float corr = expf(m - nm);
      float p = expf(s - nm);
      l = l * corr + p;
#pragma unroll
      for (int d = 0; d < 8; ++d) acc[d] = acc[d] * corr + p * vls[j * 8 + d];
      m = nm;
    }
    float invl = 1.f / l;
#pragma unroll
    for (int d = 0; d < 8; ++d) O[((long)b * L + qi) * Dm + h * DH + d] = acc[d] * invl;
    if (amean) {
      for (int j = 0; j < jend; ++j) {
        float s = 0.f;
#pragma unroll
        for (int d = 0; d < 8; ++d) s += qr[d] * kls[j * 8 + d];
        float p = expf(s * rs - m) * invl;
        atomicAdd(&amean[((long)b * L + qi) * L + j], p * invnh);
      }
    }
  }
}

__global__ void softmax16_kernel(float* __restrict__ p, int rows) {
  int r = blockIdx.x * blockDim.x + threadIdx.x;
  if (r >= rows) return;
  float* v = p + (long)r * 16;
  float m = v[0];
#pragma unroll
  for (int i = 1; i < 16; ++i) m = fmaxf(m, v[i]);
  float t[16], s = 0.f;
#pragma unroll
  for (int i = 0; i < 16; ++i) { t[i] = expf(v[i] - m); s += t[i]; }
  float inv = 1.f / s;
#pragma unroll
  for (int i = 0; i < 16; ++i) v[i] = t[i] * inv;
}

__global__ void topk_kernel(const float* __restrict__ am, float* __restrict__ out, int rows) {
  int r = blockIdx.x * blockDim.x + threadIdx.x;
  if (r >= rows) return;
  float best[8];
#pragma unroll
  for (int i = 0; i < 8; ++i) best[i] = -3.0e38f;
  for (int j = 0; j < Tc; ++j) {
    float v = am[(long)r * Tc + j];
    if (v > best[7]) {
      int i = 7;
      while (i > 0 && best[i - 1] < v) { best[i] = best[i - 1]; --i; }
      best[i] = v;
    }
  }
#pragma unroll
  for (int i = 0; i < 8; ++i) out[(long)r * 8 + i] = best[i];
}

// local hypergraph conv: one block per sequence
__global__ void local_conv_kernel(float* __restrict__ h, const float* __restrict__ Hl,
                                  const float* __restrict__ Wc, const float* __restrict__ bc) {
  int bn = blockIdx.x;
  long base = (long)bn * Tc;
  __shared__ float e[16 * 32];
  __shared__ float tmp[96 * 32];
  for (int idx = threadIdx.x; idx < 512; idx += blockDim.x) {
    int m = idx >> 5, d = idx & 31;
    float s = 0.f;
    for (int t = 0; t < Tc; ++t) s += Hl[(base + t) * 16 + m] * h[(base + t) * 32 + d];
    e[idx] = s * (1.f / Tc);
  }
  __syncthreads();
  for (int idx = threadIdx.x; idx < Tc * 32; idx += blockDim.x) {
    int t = idx >> 5, d = idx & 31;
    float s = 0.f;
#pragma unroll
    for (int m = 0; m < 16; ++m) s += Hl[(base + t) * 16 + m] * e[m * 32 + d];
    tmp[idx] = s;
  }
  __syncthreads();
  for (int idx = threadIdx.x; idx < Tc * 32; idx += blockDim.x) {
    int t = idx >> 5, d = idx & 31;
    float s = bc[d];
    for (int d2 = 0; d2 < 32; ++d2) s += tmp[t * 32 + d2] * Wc[d2 * 32 + d];
    h[(base + t) * 32 + d] += fmaxf(s, 0.f);
  }
}

// transposes for node-axis attention
__global__ void to_hn_kernel(const float* __restrict__ h, float* __restrict__ hn, int total) {
  int idx = blockIdx.x * blockDim.x + threadIdx.x;
  if (idx >= total) return;
  int d = idx & 31, r = idx >> 5;
  int n = r % Nc, bt = r / Nc, t = bt % Tc, b = bt / Tc;
  hn[idx] = h[(((long)b * Nc + n) * Tc + t) * 32 + d];
}
__global__ void from_hn_kernel(const float* __restrict__ hn, float* __restrict__ g, int total) {
  int idx = blockIdx.x * blockDim.x + threadIdx.x;
  if (idx >= total) return;
  int d = idx & 31, r = idx >> 5;  // r = (b*N+n)*T + t
  int t = r % Tc, bnn = r / Tc, n = bnn % Nc, b = bnn / Nc;
  g[idx] = hn[(((long)b * Tc + t) * Nc + n) * 32 + d];
}
__global__ void add_kernel(const float* __restrict__ a, const float* __restrict__ b,
                           float* __restrict__ c, int total) {
  int i = blockIdx.x * blockDim.x + threadIdx.x;
  if (i < total) c[i] = a[i] + b[i];
}

__global__ void wedge_kernel(const float* __restrict__ w, float* __restrict__ out) {
  int i = threadIdx.x;
  if (i < 16) { float x = w[i]; out[i] = (x > 20.f) ? x : log1pf(expf(x)); }
}

// global hypergraph: edge features (per batch), scaled by softplus(wedge)
__global__ void glob_e_kernel(const float* __restrict__ h, const float* __restrict__ Hg,
                              const float* __restrict__ wedge, float* __restrict__ eout) {
  int b = blockIdx.x;
  int idx = threadIdx.x;  // 512 threads, one (m,d) each
  int m = idx >> 5, d = idx & 31;
  long base = (long)b * NTc;
  float s = 0.f;
  for (int r = 0; r < NTc; ++r) s += Hg[(base + r) * 16 + m] * h[(base + r) * 32 + d];
  eout[b * 512 + idx] = s * (1.f / NTc) * wedge[m];
}
__global__ void glob_apply_kernel(float* __restrict__ h, const float* __restrict__ Hg,
                                  const float* __restrict__ eout, const float* __restrict__ Wc,
                                  const float* __restrict__ bc, int rows) {
  int r = blockIdx.x * blockDim.x + threadIdx.x;
  if (r >= rows) return;
  int b = r / NTc;
  float tmp[32];
  for (int d = 0; d < 32; ++d) {
    float s = 0.f;
#pragma unroll
    for (int m = 0; m < 16; ++m) s += Hg[(long)r * 16 + m] * eout[b * 512 + m * 32 + d];
    tmp[d] = s;
  }
  for (int d = 0; d < 32; ++d) {
    float s = bc[d];
    for (int d2 = 0; d2 < 32; ++d2) s += tmp[d2] * Wc[d2 * 32 + d];
    h[(long)r * 32 + d] += fmaxf(s, 0.f);
  }
}

// LayerNorm over 3072 (output head)
__global__ void ln3072_kernel(const float* __restrict__ in, float* __restrict__ out,
                              const float* __restrict__ g, const float* __restrict__ bb) {
  int row = blockIdx.x;
  __shared__ float red[256];
  const float* x = in + (long)row * TD;
  float s = 0.f;
  for (int i = threadIdx.x; i < TD; i += 256) s += x[i];
  red[threadIdx.x] = s; __syncthreads();
  for (int o = 128; o > 0; o >>= 1) { if (threadIdx.x < o) red[threadIdx.x] += red[threadIdx.x + o]; __syncthreads(); }
  float mean = red[0] / TD; __syncthreads();
  float vs = 0.f;
  for (int i = threadIdx.x; i < TD; i += 256) { float d = x[i] - mean; vs += d * d; }
  red[threadIdx.x] = vs; __syncthreads();
  for (int o = 128; o > 0; o >>= 1) { if (threadIdx.x < o) red[threadIdx.x] += red[threadIdx.x + o]; __syncthreads(); }
  float inv = rsqrtf(red[0] / TD + 1e-5f);
  for (int i = threadIdx.x; i < TD; i += 256)
    out[(long)row * TD + i] = (x[i] - mean) * inv * g[i] + bb[i];
}

// final 3072 -> 2 projection: one wave per (row, col), shuffle reduction
__global__ void out_w2_kernel(const float* __restrict__ z1, const float* __restrict__ W2,
                              const float* __restrict__ b2, float* __restrict__ z) {
  int wid = threadIdx.x >> 5, lane = threadIdx.x & 31;
  int task = blockIdx.x * (blockDim.x >> 5) + wid;
  if (task >= BNc * 2) return;
  int row = task >> 1, c = task & 1;
  float s = 0.f;
  for (int k = lane; k < TD; k += 32) s += z1[(long)row * TD + k] * W2[k * 2 + c];
  for (int o = 16; o > 0; o >>= 1) s += __shfl_xor(s, o);
  if (lane == 0) z[row * 2 + c] = s + b2[c];
}

// ---------------- host helpers ----------------
static void gemm(hipStream_t st, const float* A, int lda, const float* B, int ldb,
                 float* C, int ldc, int M, int N, int K,
                 const float* bias = nullptr, int act = 0,
                 const float* rowScale = nullptr, int rss = 1, int accum = 0) {
  // preconditions: K%32==0, N%16==0, lda%4==0 (all call sites comply)
  int ntMb = (((M + 15) / 16) + MBLK - 1) / MBLK;
  int tiles = ntMb * (N / 16);
  int blocks = (tiles + 7) / 8;
  wmma_gemm_kernel<<<blocks, 256, 0, st>>>(A, lda, B, ldb, C, ldc, M, N, K,
                                           bias, act, rowScale, rss, accum);
}
static void zero(hipStream_t st, float* p, int n) {
  zero_kernel<<<(n + 255) / 256, 256, 0, st>>>(p, n);
}

// workspace arena (floats), lifetime-based reuse
constexpr long RL = (long)Rr;
constexpr long o_h    = 0;
constexpr long o_xz   = o_h + RL * 32;
constexpr long o_am   = o_xz;                 // amean reuses xz (mamba done)
constexpr long o_dt   = o_xz + RL * 256;
constexpr long o_hid  = o_dt;                 // moe hidden reuses dt
constexpr long o_bm   = o_dt + RL * 128;
constexpr long o_cm   = o_bm + RL * 16;
constexpr long o_y    = o_cm + RL * 16;
constexpr long o_gi   = o_y;                  // gru gi reuses y
constexpr long o_tmp  = o_y + RL * 128;
constexpr long o_q    = o_tmp + RL * 32;
constexpr long o_k    = o_q + RL * 32;
constexpr long o_v    = o_k + RL * 32;
constexpr long o_o    = o_v + RL * 32;
constexpr long o_gate = o_o + RL * 32;
constexpr long o_hn   = o_gate + RL * 4;
constexpr long o_g2   = o_hn + RL * 32;
constexpr long o_sum  = o_g2 + RL * 32;
constexpr long o_z0   = o_sum + RL * 32;
constexpr long o_z1   = o_z0 + (long)BNc * TD;
constexpr long o_eg   = o_z1 + (long)BNc * TD;

static void run_mha(hipStream_t st, float* ws, float* X, int BB, int L,
                    const float* Wq, const float* Wk, const float* Wv, const float* Wo,
                    const float* g, const float* bvec, int causal, float* amean) {
  int rows = BB * L;
  gemm(st, X, 32, Wq, 32, ws + o_q, 32, rows, 32, 32);
  gemm(st, X, 32, Wk, 32, ws + o_k, 32, rows, 32, 32);
  gemm(st, X, 32, Wv, 32, ws + o_v, 32, rows, 32, 32);
  attn_kernel<<<BB * 4, 128, 0, st>>>(ws + o_q, ws + o_k, ws + o_v, ws + o_o,
                                      L, 4, causal, amean, 0.25f);
  gemm(st, ws + o_o, 32, Wo, 32, ws + o_tmp, 32, rows, 32, 32);
  add_ln32_kernel<<<(rows + 7) / 8, 256, 0, st>>>(X, ws + o_tmp, g, bvec, rows);
}

extern "C" void kernel_launch(void* const* d_in, const int* in_sizes, int n_in,
                              void* d_out, int out_size, void* d_ws, size_t ws_size,
                              hipStream_t stream) {
  (void)in_sizes; (void)n_in; (void)out_size; (void)ws_size;
  float* ws  = (float*)d_ws;
  float* out = (float*)d_out;
  auto P = [&](int i) { return (const float*)d_in[i]; };

  // embedding MLP: (B,N,T,F) -> h (BN*T, 32)
  emb_kernel<<<(Rr + 255) / 256, 256, 0, stream>>>(P(0), P(1), P(2), P(3), P(4),
                                                   ws + o_h, Rr);

  // ---- MAGE layers ----
  for (int l = 0; l < 2; ++l) {
    int mb = 5 + 29 * l;
    // mamba block
    gemm(stream, ws + o_h, 32, P(mb + 0), 256, ws + o_xz, 256, Rr, 256, 32);
    gemm(stream, ws + o_xz, 256, P(mb + 1), 128, ws + o_dt, 128, Rr, 128, 128,
         P(mb + 2), /*softplus*/2);
    gemm(stream, ws + o_xz, 256, P(mb + 3), 16, ws + o_bm, 16, Rr, 16, 128);
    gemm(stream, ws + o_xz, 256, P(mb + 4), 16, ws + o_cm, 16, Rr, 16, 128);
    mamba_scan_kernel<<<(BNc * Ec + 255) / 256, 256, 0, stream>>>(
        ws + o_xz, ws + o_dt, ws + o_bm, ws + o_cm, P(mb + 5), P(mb + 6), ws + o_y);
    gemm(stream, ws + o_y, 128, P(mb + 7), 32, ws + o_tmp, 32, Rr, 32, 128);
    add_ln32_kernel<<<(Rr + 7) / 8, 256, 0, stream>>>(ws + o_h, ws + o_tmp,
                                                      P(mb + 8), P(mb + 9), Rr);
    // MoE block
    moe_gate_kernel<<<(Rr + 255) / 256, 256, 0, stream>>>(ws + o_h, P(mb + 10),
                                                          ws + o_gate, Rr);
    for (int e = 0; e < 4; ++e) {
      gemm(stream, ws + o_h, 32, P(mb + 11) + (long)e * 32 * 128, 128,
           ws + o_hid, 128, Rr, 128, 32, P(mb + 12) + e * 128, /*relu*/1);
      gemm(stream, ws + o_hid, 128, P(mb + 13) + (long)e * 128 * 32, 32,
           ws + o_tmp, 32, Rr, 32, 128, P(mb + 14) + e * 32, 0,
           ws + o_gate + e, 4, e > 0);
    }
    add_ln32_kernel<<<(Rr + 7) / 8, 256, 0, stream>>>(ws + o_h, ws + o_tmp,
                                                      P(mb + 15), P(mb + 16), Rr);
    // GRU block
    gemm(stream, ws + o_h, 32, P(mb + 17), 96, ws + o_gi, 96, Rr, 96, 32,
         P(mb + 19), 0);
    gru_scan_kernel<<<BNc, 32, 0, stream>>>(ws + o_gi, P(mb + 18), P(mb + 20),
                                            ws + o_tmp);
    add_ln32_kernel<<<(Rr + 7) / 8, 256, 0, stream>>>(ws + o_h, ws + o_tmp,
                                                      P(mb + 21), P(mb + 22), Rr);
    // MHA block
    run_mha(stream, ws, ws + o_h, BNc, Tc, P(mb + 23), P(mb + 24), P(mb + 25),
            P(mb + 26), P(mb + 27), P(mb + 28), 0, nullptr);
  }

  // ---- f2d attention (writes att_F = mean over heads) ----
  zero(stream, out + OATT, Rr * Tc);
  run_mha(stream, ws, ws + o_h, BNc, Tc, P(63), P(64), P(65), P(66), P(67), P(68),
          0, out + OATT);

  // ---- local hypergraph: causal MHA + topK + incidence + conv ----
  zero(stream, ws + o_am, Rr * Tc);
  run_mha(stream, ws, ws + o_h, BNc, Tc, P(69), P(70), P(71), P(72), P(73), P(74),
          1, ws + o_am);
  topk_kernel<<<(Rr + 255) / 256, 256, 0, stream>>>(ws + o_am, out + OTOPK, Rr);
  gemm(stream, ws + o_h, 32, P(75), 16, out + OHLOC, 16, Rr, 16, 32);
  softmax16_kernel<<<(Rr + 255) / 256, 256, 0, stream>>>(out + OHLOC, Rr);
  local_conv_kernel<<<BNc, 256, 0, stream>>>(ws + o_h, out + OHLOC, P(76), P(77));

  // ---- s2d node-axis attention ----
  to_hn_kernel<<<(Rr * 32 + 255) / 256, 256, 0, stream>>>(ws + o_h, ws + o_hn, Rr * 32);
  run_mha(stream, ws, ws + o_hn, Bc * Tc, Nc, P(78), P(79), P(80), P(81), P(82), P(83),
          0, nullptr);
  from_hn_kernel<<<(Rr * 32 + 255) / 256, 256, 0, stream>>>(ws + o_hn, ws + o_g2, Rr * 32);
  add_kernel<<<(Rr * 32 + 255) / 256, 256, 0, stream>>>(ws + o_h, ws + o_g2,
                                                        ws + o_sum, Rr * 32);

  // ---- global hypergraph ----
  gemm(stream, ws + o_sum, 32, P(84), 16, out + OHG, 16, Rr, 16, 32);
  softmax16_kernel<<<(Rr + 255) / 256, 256, 0, stream>>>(out + OHG, Rr);
  wedge_kernel<<<1, 32, 0, stream>>>(P(85), out + OWEDGE);
  glob_e_kernel<<<Bc, 512, 0, stream>>>(ws + o_h, out + OHG, out + OWEDGE, ws + o_eg);
  glob_apply_kernel<<<(Rr + 255) / 256, 256, 0, stream>>>(ws + o_h, out + OHG,
                                                          ws + o_eg, P(86), P(87), Rr);

  // ---- output head ----
  ln3072_kernel<<<BNc, 256, 0, stream>>>(ws + o_h, ws + o_z0, P(88), P(89));
  gemm(stream, ws + o_z0, TD, P(90), TD, ws + o_z1, TD, BNc, TD, TD, P(91), /*relu*/1);
  out_w2_kernel<<<(BNc * 2 + 7) / 8, 256, 0, stream>>>(ws + o_z1, P(92), P(93),
                                                       out + OZ);
}
